// GATConv_67671504716395
// MI455X (gfx1250) — compile-verified, matrix operand order checked by reading, not run on hardware
//
#include <hip/hip_runtime.h>
#include <hip/hip_bf16.h>
#include <math.h>

// ---------------- constants from the reference ----------------
#define FN_IN   256   // F_IN, also H*D
#define F_EDGE  64
#define NHEAD   4
#define HDIM    64
#define NEG_SLOPE 0.2f

typedef __attribute__((ext_vector_type(16))) _Float16 v16h;
typedef __attribute__((ext_vector_type(8)))  float    v8f;

// order-preserving float -> uint key for atomic max
__device__ __forceinline__ unsigned fkey(float f) {
    unsigned b = __float_as_uint(f);
    return (b & 0x80000000u) ? ~b : (b | 0x80000000u);
}
__device__ __forceinline__ float fkey_dec(unsigned k) {
    unsigned b = (k & 0x80000000u) ? (k & 0x7FFFFFFFu) : ~k;
    return __uint_as_float(b);
}

// WMMA 16-bit A-fragment K permutation within a 32-group:
// fragment-contiguous order is [0..7 | 16..23 | 8..15 | 24..31]
__device__ __forceinline__ int kperm(int p) {
    int q   = p & 7;
    int blk = p >> 3;                                   // 0..3
    int base = ((blk & 1) << 4) | ((blk & 2) << 2);     // 0,16,8,24
    return base + q;
}

// ---------------- 0) init scratch / output --------------------
__global__ void init_kernel(float* __restrict__ out, float* __restrict__ denom,
                            float* __restrict__ deg, unsigned* __restrict__ mkey,
                            int N) {
    int tid = blockIdx.x * blockDim.x + threadIdx.x;
    int total = N * FN_IN;
    if (tid < total) out[tid] = 0.0f;
    if (tid < N * NHEAD) { denom[tid] = 0.0f; mkey[tid] = 0u; }
    if (tid < N) deg[tid] = 0.0f;
}

// ---------------- 1) V[f,h] = sum_d W_edge[h*64+d, f] * attn_e[h*64+d]
__global__ void ve_kernel(const float* __restrict__ Wedge,
                          const float* __restrict__ attn_e,
                          float* __restrict__ Ve) {
    int t = threadIdx.x;            // 256 threads, one per (f,h)
    int f = t >> 2, h = t & 3;
    float s = 0.0f;
    #pragma unroll 8
    for (int d = 0; d < HDIM; ++d)
        s += Wedge[(size_t)(h * HDIM + d) * F_EDGE + f] * attn_e[h * HDIM + d];
    Ve[f * NHEAD + h] = s;
}

// ---------------- 2a) pre-convert h -> f16, fragment-swizzled K order
__global__ void convert_h_kernel(const float* __restrict__ H,
                                 _Float16* __restrict__ h16, int N) {
    int tid = blockIdx.x * blockDim.x + threadIdx.x;
    if (tid >= N * FN_IN) return;
    int row = tid >> 8, j = tid & 255;
    int g = j >> 5, p = j & 31;
    int k = g * 32 + kperm(p);
    h16[tid] = (_Float16)H[(size_t)row * FN_IN + k];
}

// ---------------- 2b) pre-convert [Wfc;Wres] -> f16, same swizzle
__global__ void convert_w_kernel(const float* __restrict__ Wfc,
                                 const float* __restrict__ Wres,
                                 _Float16* __restrict__ w16) {
    int tid = blockIdx.x * blockDim.x + threadIdx.x;   // 512*256 threads
    if (tid >= 2 * FN_IN * FN_IN) return;
    int row = tid >> 8, j = tid & 255;
    int g = j >> 5, p = j & 31;
    int k = g * 32 + kperm(p);
    const float* W = (row < FN_IN) ? (Wfc  + (size_t)row * FN_IN)
                                   : (Wres + (size_t)(row - FN_IN) * FN_IN);
    w16[tid] = (_Float16)W[k];
}

// ---------------- 2c) WMMA node GEMM: feat = h@Wfc^T, res = h@Wres^T
// block = 256 threads (8 waves). Block computes rows [row0,row0+16) x 512 cols.
// Operands are pre-converted f16 in fragment-contiguous K order:
// a lane's 16-element fragment half is one 32-byte contiguous chunk.
__global__ __launch_bounds__(256) void node_gemm_wmma(
    const _Float16* __restrict__ h16, const _Float16* __restrict__ w16,
    float* __restrict__ feat, float* __restrict__ res, int N)
{
    const int lane = threadIdx.x & 31;
    const int wave = threadIdx.x >> 5;
    const int row0 = blockIdx.x * 16;
    const int half = lane >> 4;          // selects K half of the fragment
    const int l16  = lane & 15;

    int m = row0 + l16;                  // A-matrix row held by this lane
    if (m >= N) m = N - 1;               // clamp (N is a multiple of 16 anyway)

    const _Float16* ha = h16 + (size_t)m * FN_IN + half * 16;

    v8f acc[4] = {};

    #pragma unroll
    for (int kt = 0; kt < 8; ++kt) {     // K = 256 in steps of 32
        v16h a = *(const v16h*)(ha + kt * 32);
        #pragma unroll
        for (int t = 0; t < 4; ++t) {
            const int col = wave * 64 + t * 16 + l16;   // 0..511
            v16h b = *(const v16h*)(w16 + (size_t)col * FN_IN + kt * 32 + half * 16);
            acc[t] = __builtin_amdgcn_wmma_f32_16x16x32_f16(
                false, a, false, b, (short)0, acc[t], false, false);
        }
    }

    // C/D layout: lane(0-15): VGPR r -> M=r,N=lane ; lane(16-31): M=r+8,N=lane-16
    #pragma unroll
    for (int t = 0; t < 4; ++t) {
        const int colb = wave * 64 + t * 16;
        #pragma unroll
        for (int r = 0; r < 8; ++r) {
            const int mm = row0 + half * 8 + r;
            const int cc = colb + l16;
            if (mm < N) {
                float v = acc[t][r];
                if (cc < FN_IN) feat[(size_t)mm * FN_IN + cc]           = v;
                else            res [(size_t)mm * FN_IN + (cc - FN_IN)] = v;
            }
        }
    }
}

// ---------------- 3) el/er per (node, head) -------------------
__global__ void elr_kernel(const float* __restrict__ feat,
                           const float* __restrict__ attn_l,
                           const float* __restrict__ attn_r,
                           float* __restrict__ el, float* __restrict__ er, int N) {
    int tid = blockIdx.x * blockDim.x + threadIdx.x;
    if (tid >= N * NHEAD) return;
    int n = tid >> 2, h = tid & 3;
    const float* fp = feat + (size_t)n * FN_IN + h * HDIM;
    float sl = 0.0f, sr = 0.0f;
    #pragma unroll 8
    for (int d = 0; d < HDIM; ++d) {
        float f = fp[d];
        sl += f * attn_l[h * HDIM + d];
        sr += f * attn_r[h * HDIM + d];
    }
    el[tid] = sl;
    er[tid] = sr;
}

// ---------------- 4) edge logits + leaky relu + seg-max + degree
__global__ void edge_score_kernel(const float* __restrict__ ef,
                                  const float* __restrict__ Ve,
                                  const float* __restrict__ el,
                                  const float* __restrict__ er,
                                  const int* __restrict__ src,
                                  const int* __restrict__ dst,
                                  float* __restrict__ escore,
                                  unsigned* __restrict__ mkey,
                                  float* __restrict__ deg, int E) {
    int tid = blockIdx.x * blockDim.x + threadIdx.x;
    if (tid >= E * NHEAD) return;
    int e = tid >> 2, h = tid & 3;
    const float* ep = ef + (size_t)e * F_EDGE;
    float s = 0.0f;
    #pragma unroll 8
    for (int f = 0; f < F_EDGE; ++f) s += ep[f] * Ve[f * NHEAD + h];
    int sn = src[e], dn = dst[e];
    float v = el[sn * NHEAD + h] + er[dn * NHEAD + h] + s;
    v = (v > 0.0f) ? v : NEG_SLOPE * v;
    escore[tid] = v;
    atomicMax(&mkey[dn * NHEAD + h], fkey(v));
    if (h == 0) atomicAdd(&deg[dn], 1.0f);
}

// ---------------- 5) exp + denominator ------------------------
__global__ void edge_exp_kernel(const int* __restrict__ dst,
                                const unsigned* __restrict__ mkey,
                                float* __restrict__ escore,
                                float* __restrict__ denom, int E) {
    int tid = blockIdx.x * blockDim.x + threadIdx.x;
    if (tid >= E * NHEAD) return;
    int e = tid >> 2, h = tid & 3;
    int dn = dst[e];
    float m = fkey_dec(mkey[dn * NHEAD + h]);
    float x = __expf(escore[tid] - m);
    escore[tid] = x;                       // now holds exp(e - m)
    atomicAdd(&denom[dn * NHEAD + h], x);
}

// ---------------- 6) message aggregation ----------------------
// one 256-thread block per edge; thread t handles feature component t.
__global__ __launch_bounds__(256) void aggregate_kernel(
    const float* __restrict__ feat, const float* __restrict__ escore,
    const float* __restrict__ denom, const int* __restrict__ src,
    const int* __restrict__ dst, float* __restrict__ out, int E) {
    int e = blockIdx.x;
    if (e >= E) return;
    int t = threadIdx.x;
    int h = t >> 6;                        // head = t / 64
    int sn = src[e], dn = dst[e];
    float a = escore[e * NHEAD + h] / denom[dn * NHEAD + h];
    float msg = feat[(size_t)sn * FN_IN + t] * a;
    atomicAdd(&out[(size_t)dn * FN_IN + t], msg);
}

// ---------------- 7) finalize: out = out/deg + res ------------
__global__ void finalize_kernel(float* __restrict__ out,
                                const float* __restrict__ res,
                                const float* __restrict__ deg, int N) {
    int tid = blockIdx.x * blockDim.x + threadIdx.x;
    if (tid >= N * FN_IN) return;
    int n = tid >> 8;
    float dg = fmaxf(deg[n], 1.0f);
    out[tid] = out[tid] / dg + res[tid];
}

// ------------------------------------------------------------------
extern "C" void kernel_launch(void* const* d_in, const int* in_sizes, int n_in,
                              void* d_out, int out_size, void* d_ws, size_t ws_size,
                              hipStream_t stream) {
    const float* h         = (const float*)d_in[0];
    const float* edge_feat = (const float*)d_in[1];
    const int*   src       = (const int*)  d_in[2];
    const int*   dst       = (const int*)  d_in[3];
    const float* Wfc       = (const float*)d_in[4];
    const float* Wedge     = (const float*)d_in[5];
    const float* Wres      = (const float*)d_in[6];
    const float* attn_l    = (const float*)d_in[7];
    const float* attn_r    = (const float*)d_in[8];
    const float* attn_e    = (const float*)d_in[9];

    const int N = in_sizes[0] / FN_IN;    // 50000
    const int E = in_sizes[2];            // 800000
    float* out = (float*)d_out;

    // ---- carve workspace (256B aligned chunks) ----
    char* w = (char*)d_ws;
    auto carve = [&](size_t bytes) -> char* {
        char* p = w;
        w += (bytes + 255) & ~(size_t)255;
        return p;
    };
    float*     feat   = (float*)    carve((size_t)N * FN_IN * sizeof(float));
    float*     res    = (float*)    carve((size_t)N * FN_IN * sizeof(float));
    _Float16*  h16    = (_Float16*) carve((size_t)N * FN_IN * sizeof(_Float16));
    _Float16*  w16    = (_Float16*) carve((size_t)2 * FN_IN * FN_IN * sizeof(_Float16));
    float*     el     = (float*)    carve((size_t)N * NHEAD * sizeof(float));
    float*     er     = (float*)    carve((size_t)N * NHEAD * sizeof(float));
    float*     escore = (float*)    carve((size_t)E * NHEAD * sizeof(float));
    float*     denom  = (float*)    carve((size_t)N * NHEAD * sizeof(float));
    float*     deg    = (float*)    carve((size_t)N * sizeof(float));
    unsigned*  mkey   = (unsigned*) carve((size_t)N * NHEAD * sizeof(unsigned));
    float*     Ve     = (float*)    carve((size_t)F_EDGE * NHEAD * sizeof(float));

    const int TB = 256;

    // 0) init
    init_kernel<<<(N * FN_IN + TB - 1) / TB, TB, 0, stream>>>(out, denom, deg, mkey, N);
    // 1) collapsed edge-attention matrix V
    ve_kernel<<<1, 256, 0, stream>>>(Wedge, attn_e, Ve);
    // 2a/2b) one-time f16 conversion + fragment swizzle of GEMM operands
    convert_h_kernel<<<(N * FN_IN + TB - 1) / TB, TB, 0, stream>>>(h, h16, N);
    convert_w_kernel<<<(2 * FN_IN * FN_IN + TB - 1) / TB, TB, 0, stream>>>(Wfc, Wres, w16);
    // 2c) WMMA GEMM: feat & res
    node_gemm_wmma<<<(N + 15) / 16, 256, 0, stream>>>(h16, w16, feat, res, N);
    // 3) el / er
    elr_kernel<<<(N * NHEAD + TB - 1) / TB, TB, 0, stream>>>(feat, attn_l, attn_r, el, er, N);
    // 4) edge logits + seg-max + degree
    edge_score_kernel<<<(E * NHEAD + TB - 1) / TB, TB, 0, stream>>>(
        edge_feat, Ve, el, er, src, dst, escore, mkey, deg, E);
    // 5) exp + denominator
    edge_exp_kernel<<<(E * NHEAD + TB - 1) / TB, TB, 0, stream>>>(dst, mkey, escore, denom, E);
    // 6) aggregation (one block per edge)
    aggregate_kernel<<<E, 256, 0, stream>>>(feat, escore, denom, src, dst, out, E);
    // 7) finalize
    finalize_kernel<<<(N * FN_IN + TB - 1) / TB, TB, 0, stream>>>(out, res, deg, N);
}
